// spatial_encoder_61607010894235
// MI455X (gfx1250) — compile-verified, matrix operand order checked by reading, not run on hardware
//
#include <hip/hip_runtime.h>
#include <hip/hip_bf16.h>
#include <stdint.h>

#define D_MODEL 1936
#define DFF     2048
#define NCLS    26
#define NHEAD   8
#define DH      242      // D_MODEL / NHEAD
#define FRAMES  1000
#define LMAX    16       // bbox_num (max of counts = 8..16)
#define MPAD    (FRAMES*LMAX)

typedef __bf16 bf16_t;
typedef __attribute__((ext_vector_type(16))) __bf16 v16bf;
typedef __attribute__((ext_vector_type(8)))  float  v8f;
typedef __attribute__((ext_vector_type(4)))  float  f32x4;

// Pack two fp32 -> two bf16 (RNE) in one u32.
// 32-bit-only dataflow: v_bfe/v_and + v_add3 per value, then one v_perm_b32
// selecting the high halves: result = { b[31:16], a[31:16] }.
__device__ __forceinline__ unsigned pack2bf(float a, float b) {
  unsigned ua = __builtin_bit_cast(unsigned, a);
  unsigned ub = __builtin_bit_cast(unsigned, b);
  ua += 0x7FFFu + ((ua >> 16) & 1u);
  ub += 0x7FFFu + ((ub >> 16) & 1u);
  return __builtin_amdgcn_perm(ub, ua, 0x07060302u);
}

// ---------------------------------------------------------------------------
// Generic tiled GEMM:  C[M,N] = (A [+ A2]) (M,K)  x  W(N,K)^T  + bias, opt ReLU
// bf16 WMMA 16x16x32, fp32 accumulate. Block 256 threads = 8 waves (2x4),
// block tile 128x128, wave tile 64x32 (4x2 WMMA frags), BK = 32.
// Register-prefetch pipelined: next tile's global loads overlap current WMMAs.
// ---------------------------------------------------------------------------
#define BM 128
#define BN 128
#define BK 32
#define LSA 40   // LDS row stride in bf16 elems (80 B: 16B-aligned rows)

template<bool HAS_A2>
__global__ __launch_bounds__(256)
void gemm_bf16_wmma(const float* __restrict__ A, const float* __restrict__ A2,
                    const float* __restrict__ W, const float* __restrict__ bias,
                    float* __restrict__ C, int M, int N, int K, int relu)
{
  __shared__ __align__(16) bf16_t As[BM * LSA];
  __shared__ __align__(16) bf16_t Bs[BN * LSA];

  const int tid  = threadIdx.x;
  const int bm0  = blockIdx.y * BM;
  const int bn0  = blockIdx.x * BN;
  const int wid  = tid >> 5;
  const int lane = tid & 31;
  const int wm   = wid >> 2;          // 0..1  (64 rows each)
  const int wn   = wid & 3;           // 0..3  (32 cols each)
  const int lr   = lane & 15;
  const int sel  = lane >> 4;

  const bool mFull = (bm0 + BM <= M);
  const bool nFull = (bn0 + BN <= N);
  const bool k4    = ((K & 3) == 0);

  // staging: tile = 128x32 floats = 1024 float4 chunks; 4 per thread
  // chunk idx = tid + i*256 ; row = idx>>3 ; col = (idx&7)*4
  f32x4 aReg[4], wReg[4];

  auto loadA = [&](int k0) {
    const bool fast = mFull && k4 && (k0 + BK <= K);
#pragma unroll
    for (int i = 0; i < 4; ++i) {
      int idx = tid + i * 256;
      int r = idx >> 3, c = (idx & 7) * 4;
      int gm = bm0 + r, gk = k0 + c;
      if (fast) {
        size_t off = (size_t)gm * K + gk;
        f32x4 v = *reinterpret_cast<const f32x4*>(A + off);
        if (HAS_A2) {
          f32x4 v2 = *reinterpret_cast<const f32x4*>(A2 + off);
          v += v2;
        }
        aReg[i] = v;
      } else {
        f32x4 v = {0.f, 0.f, 0.f, 0.f};
        if (gm < M) {
#pragma unroll
          for (int e = 0; e < 4; ++e)
            if (gk + e < K) {
              size_t off = (size_t)gm * K + gk + e;
              float t = A[off];
              if (HAS_A2) t += A2[off];
              v[e] = t;
            }
        }
        aReg[i] = v;
      }
    }
  };

  auto loadW = [&](int k0) {
    const bool fast = nFull && k4 && (k0 + BK <= K);
#pragma unroll
    for (int i = 0; i < 4; ++i) {
      int idx = tid + i * 256;
      int r = idx >> 3, c = (idx & 7) * 4;
      int gn = bn0 + r, gk = k0 + c;
      if (fast) {
        wReg[i] = *reinterpret_cast<const f32x4*>(W + (size_t)gn * K + gk);
      } else {
        f32x4 v = {0.f, 0.f, 0.f, 0.f};
        if (gn < N) {
#pragma unroll
          for (int e = 0; e < 4; ++e)
            if (gk + e < K) v[e] = W[(size_t)gn * K + gk + e];
        }
        wReg[i] = v;
      }
    }
  };

  auto storeTiles = [&]() {
#pragma unroll
    for (int i = 0; i < 4; ++i) {
      int idx = tid + i * 256;
      int r = idx >> 3, c = (idx & 7) * 4;
      uint2 qa = make_uint2(pack2bf(aReg[i][0], aReg[i][1]),
                            pack2bf(aReg[i][2], aReg[i][3]));
      uint2 qw = make_uint2(pack2bf(wReg[i][0], wReg[i][1]),
                            pack2bf(wReg[i][2], wReg[i][3]));
      *reinterpret_cast<uint2*>(&As[r * LSA + c]) = qa;
      *reinterpret_cast<uint2*>(&Bs[r * LSA + c]) = qw;
    }
  };

  v8f acc[4][2] = {};

  const int ksteps = (K + BK - 1) / BK;
  loadA(0);
  loadW(0);
  for (int kt = 0; kt < ksteps; ++kt) {
    __syncthreads();             // prior compute finished reading LDS
    storeTiles();
    __syncthreads();
    if (kt + 1 < ksteps) {       // prefetch next tile; overlaps WMMAs below
      loadA((kt + 1) * BK);
      loadW((kt + 1) * BK);
    }

    // Fragments per documented 16-bit A/B VGPR layout:
    // lanes 0-15: K0..7 | K16..23 ; lanes 16-31: K8..15 | K24..31
    v16bf afrag[4], bfrag[2];
#pragma unroll
    for (int mi = 0; mi < 4; ++mi) {
      const bf16_t* ap = &As[(wm * 64 + mi * 16 + lr) * LSA];
      union { f32x4 f[2]; v16bf v; } u;
      u.f[0] = *reinterpret_cast<const f32x4*>(ap + sel * 8);
      u.f[1] = *reinterpret_cast<const f32x4*>(ap + 16 + sel * 8);
      afrag[mi] = u.v;
    }
#pragma unroll
    for (int ni = 0; ni < 2; ++ni) {
      const bf16_t* bp = &Bs[(wn * 32 + ni * 16 + lr) * LSA];
      union { f32x4 f[2]; v16bf v; } u;
      u.f[0] = *reinterpret_cast<const f32x4*>(bp + sel * 8);
      u.f[1] = *reinterpret_cast<const f32x4*>(bp + 16 + sel * 8);
      bfrag[ni] = u.v;
    }

#pragma unroll
    for (int mi = 0; mi < 4; ++mi)
#pragma unroll
      for (int ni = 0; ni < 2; ++ni)
        acc[mi][ni] = __builtin_amdgcn_wmma_f32_16x16x32_bf16(
            false, afrag[mi], false, bfrag[ni], (short)0, acc[mi][ni],
            false, false);
  }

  // epilogue: D layout -> VGPR r: lanes0-15 M=r, lanes16-31 M=r+8
  if (mFull && nFull) {
#pragma unroll
    for (int mi = 0; mi < 4; ++mi)
#pragma unroll
      for (int ni = 0; ni < 2; ++ni) {
        int gn = bn0 + wn * 32 + ni * 16 + lr;
        float bv = bias ? bias[gn] : 0.f;
#pragma unroll
        for (int r = 0; r < 8; ++r) {
          int gm = bm0 + wm * 64 + mi * 16 + sel * 8 + r;
          float o = acc[mi][ni][r] + bv;
          if (relu) o = fmaxf(o, 0.f);
          C[(size_t)gm * N + gn] = o;
        }
      }
  } else {
#pragma unroll
    for (int mi = 0; mi < 4; ++mi)
#pragma unroll
      for (int ni = 0; ni < 2; ++ni) {
        int gn = bn0 + wn * 32 + ni * 16 + lr;
        float bv = (bias && gn < N) ? bias[gn] : 0.f;
#pragma unroll
        for (int r = 0; r < 8; ++r) {
          int gm = bm0 + wm * 64 + mi * 16 + sel * 8 + r;
          if (gm < M && gn < N) {
            float o = acc[mi][ni][r] + bv;
            if (relu) o = fmaxf(o, 0.f);
            C[(size_t)gm * N + gn] = o;
          }
        }
      }
  }
}

// ---------------------------------------------------------------------------
// Meta: per-frame counts + exclusive starts; pos[i] = i - starts[frame]
// ---------------------------------------------------------------------------
__global__ void meta_kernel(const int* __restrict__ im_idx, int N,
                            int* __restrict__ counts, int* __restrict__ starts)
{
  int tid = threadIdx.x;
  for (int f = tid; f < FRAMES; f += blockDim.x) counts[f] = 0;
  __syncthreads();
  for (int i = tid; i < N; i += blockDim.x) atomicAdd(&counts[im_idx[i]], 1);
  __syncthreads();
  if (tid == 0) {
    int acc = 0;
    for (int f = 0; f < FRAMES; ++f) { starts[f] = acc; acc += counts[f]; }
  }
}

__global__ void pos_kernel(const int* __restrict__ im_idx,
                           const int* __restrict__ starts,
                           int* __restrict__ pos, int N)
{
  int i = blockIdx.x * blockDim.x + threadIdx.x;
  if (i < N) pos[i] = i - starts[im_idx[i]];
}

__global__ void prior_gather_kernel(const float* __restrict__ initialPrior,
                                    const int* __restrict__ pair_idx,
                                    const int* __restrict__ labels,
                                    float* __restrict__ p0, int N)
{
  int i = blockIdx.x;
  int lab = labels[pair_idx[2 * i + 1]];
  for (int c = threadIdx.x; c < NCLS; c += blockDim.x)
    p0[(size_t)i * NCLS + c] = initialPrior[lab * NCLS + c];
}

__global__ __launch_bounds__(256)
void scatter_kernel(const float* __restrict__ feat, const float* __restrict__ priors,
                    const int* __restrict__ im_idx, const int* __restrict__ pos,
                    float* __restrict__ X, float* __restrict__ P, int N)
{
  int i = blockIdx.x;
  int row = im_idx[i] * LMAX + pos[i];
  size_t dst = (size_t)row * D_MODEL;
  size_t src = (size_t)i * D_MODEL;
  for (int d = threadIdx.x; d < D_MODEL; d += 256) {
    X[dst + d] = feat[src + d];
    P[dst + d] = priors[src + d];
  }
}

// ---------------------------------------------------------------------------
// Attention: one block per (frame, head). L=16, dh=242, masked softmax.
// ---------------------------------------------------------------------------
__global__ __launch_bounds__(256)
void attention_kernel(const float* __restrict__ Q, const float* __restrict__ K,
                      const float* __restrict__ V, const int* __restrict__ counts,
                      float* __restrict__ Out)
{
  __shared__ float qs[LMAX * DH];
  __shared__ float ks[LMAX * DH];
  __shared__ float vs[LMAX * DH];
  __shared__ float aw[LMAX * LMAX];

  int f = blockIdx.x, h = blockIdx.y, tid = threadIdx.x;
  int cnt = counts[f];
  size_t base = (size_t)f * LMAX * D_MODEL + (size_t)h * DH;

  for (int idx = tid; idx < LMAX * DH; idx += 256) {
    int l = idx / DH, j = idx % DH;
    size_t g = base + (size_t)l * D_MODEL + j;
    qs[idx] = Q[g]; ks[idx] = K[g]; vs[idx] = V[g];
  }
  __syncthreads();

  // 256 threads = 16x16 score matrix
  int l = tid >> 4, m = tid & 15;
  const float* qp = &qs[l * DH];
  const float* kp = &ks[m * DH];
  float s = 0.f;
  for (int j = 0; j < DH; ++j) s += qp[j] * kp[j];
  s *= rsqrtf((float)DH);
  if (m >= cnt) s = -__builtin_inff();

  // softmax over m: 16-lane reductions (xor masks stay within half-wave)
  float mx = s;
  for (int off = 8; off; off >>= 1) mx = fmaxf(mx, __shfl_xor(mx, off, 32));
  float p = __expf(s - mx);
  float sum = p;
  for (int off = 8; off; off >>= 1) sum += __shfl_xor(sum, off, 32);
  aw[l * 16 + m] = p / sum;
  __syncthreads();

  for (int idx = tid; idx < LMAX * DH; idx += 256) {
    int ll = idx / DH, j = idx % DH;
    float acc = 0.f;
#pragma unroll
    for (int mm = 0; mm < LMAX; ++mm) acc += aw[ll * 16 + mm] * vs[mm * DH + j];
    Out[base + (size_t)ll * D_MODEL + j] = acc;
  }
}

// ---------------------------------------------------------------------------
// Residual + LayerNorm, in place on X:  X = LN(X + Y) * g + b
// ---------------------------------------------------------------------------
__global__ __launch_bounds__(256)
void resid_ln_kernel(float* __restrict__ X, const float* __restrict__ Y,
                     const float* __restrict__ g, const float* __restrict__ b)
{
  __shared__ float buf[D_MODEL];
  __shared__ float red[2][8];
  int row = blockIdx.x, tid = threadIdx.x;
  size_t base = (size_t)row * D_MODEL;

  float s = 0.f, s2 = 0.f;
  for (int j = tid; j < D_MODEL; j += 256) {
    float t = X[base + j] + Y[base + j];
    buf[j] = t; s += t; s2 += t * t;
  }
  for (int off = 16; off; off >>= 1) {
    s  += __shfl_xor(s, off, 32);
    s2 += __shfl_xor(s2, off, 32);
  }
  int wid = tid >> 5;
  if ((tid & 31) == 0) { red[0][wid] = s; red[1][wid] = s2; }
  __syncthreads();
  float ts = 0.f, ts2 = 0.f;
#pragma unroll
  for (int w = 0; w < 8; ++w) { ts += red[0][w]; ts2 += red[1][w]; }
  float mean = ts / D_MODEL;
  float var  = ts2 / D_MODEL - mean * mean;
  float inv  = rsqrtf(var + 1e-5f);
  for (int j = tid; j < D_MODEL; j += 256)
    X[base + j] = (buf[j] - mean) * inv * g[j] + b[j];
}

__global__ __launch_bounds__(256)
void gather_kernel(const float* __restrict__ Xf, const float* __restrict__ P,
                   const int* __restrict__ im_idx, const int* __restrict__ pos,
                   float* __restrict__ out, int N)
{
  int i = blockIdx.x;
  int row = im_idx[i] * LMAX + pos[i];
  size_t src = (size_t)row * D_MODEL;
  size_t d0 = (size_t)i * D_MODEL;
  size_t d1 = (size_t)N * D_MODEL + d0;
  for (int d = threadIdx.x; d < D_MODEL; d += 256) {
    out[d0 + d] = Xf[src + d];
    out[d1 + d] = P[src + d];
  }
}

// ---------------------------------------------------------------------------
extern "C" void kernel_launch(void* const* d_in, const int* in_sizes, int n_in,
                              void* d_out, int out_size, void* d_ws, size_t ws_size,
                              hipStream_t stream)
{
  // inputs in setup_inputs() dict order; params flattened in dict order
  const float* features     = (const float*)d_in[0];
  const int*   im_idx       = (const int*)  d_in[1];
  const int*   pair_idx     = (const int*)  d_in[2];
  const int*   labels       = (const int*)  d_in[3];
  const float* initialPrior = (const float*)d_in[4];
  const float* fc1_w = (const float*)d_in[5];
  const float* fc1_b = (const float*)d_in[6];
  const float* fc2_w = (const float*)d_in[7];
  const float* fc2_b = (const float*)d_in[8];
  const float* fc3_w = (const float*)d_in[9];
  const float* fc3_b = (const float*)d_in[10];
  const float* fc4_w = (const float*)d_in[11];
  const float* fc4_b = (const float*)d_in[12];
  const float* Wq_w  = (const float*)d_in[13];
  const float* Wq_b  = (const float*)d_in[14];
  const float* Wk_w  = (const float*)d_in[15];
  const float* Wk_b  = (const float*)d_in[16];
  const float* Wv_w  = (const float*)d_in[17];
  const float* Wv_b  = (const float*)d_in[18];
  const float* in_proj_w  = (const float*)d_in[19];
  const float* in_proj_b  = (const float*)d_in[20];
  const float* out_proj_w = (const float*)d_in[21];
  const float* out_proj_b = (const float*)d_in[22];
  const float* lin1_w = (const float*)d_in[23];
  const float* lin1_b = (const float*)d_in[24];
  const float* lin2_w = (const float*)d_in[25];
  const float* lin2_b = (const float*)d_in[26];
  const float* ln1_g  = (const float*)d_in[27];
  const float* ln1_b  = (const float*)d_in[28];
  const float* ln2_g  = (const float*)d_in[29];
  const float* ln2_b  = (const float*)d_in[30];

  const int N = in_sizes[1];   // number of boxes (im_idx length)

  // -------- workspace layout --------
  char* ws = (char*)d_ws;
  const size_t szMat = (size_t)MPAD * D_MODEL * sizeof(float);
  float* X  = (float*)(ws + 0 * szMat);   // padded src (and LN outputs)
  float* P  = (float*)(ws + 1 * szMat);   // padded priors
  float* Q  = (float*)(ws + 2 * szMat);
  float* Kb = (float*)(ws + 3 * szMat);
  float* V  = (float*)(ws + 4 * szMat);
  float* T1 = (float*)(ws + 5 * szMat);
  float* T2 = (float*)(ws + 6 * szMat);
  float* Hf = (float*)(ws + 7 * szMat);   // MPAD x DFF
  char* meta = ws + 7 * szMat + (size_t)MPAD * DFF * sizeof(float);
  int* counts = (int*)meta;
  int* starts = counts + FRAMES;
  int* pos    = starts + FRAMES;

  // small prior-MLP activations reuse large buffers (stream-ordered, safe)
  float* p0      = Q;    // N x 26
  float* h1      = Kb;   // N x 128
  float* h2      = V;    // N x 256
  float* h3      = T2;   // N x 512
  float* priorsN = T1;   // N x 1936

  auto gemm = [&](const float* A, const float* A2, const float* W,
                  const float* bias, float* C, int M, int Nc, int K, int relu) {
    dim3 grid((Nc + BN - 1) / BN, (M + BM - 1) / BM);
    if (A2)
      gemm_bf16_wmma<true><<<grid, 256, 0, stream>>>(A, A2, W, bias, C, M, Nc, K, relu);
    else
      gemm_bf16_wmma<false><<<grid, 256, 0, stream>>>(A, nullptr, W, bias, C, M, Nc, K, relu);
  };

  // 1) metadata
  meta_kernel<<<1, 1024, 0, stream>>>(im_idx, N, counts, starts);
  pos_kernel<<<(N + 255) / 256, 256, 0, stream>>>(im_idx, starts, pos, N);

  // 2) prior MLP (WMMA GEMM chain)
  prior_gather_kernel<<<N, 32, 0, stream>>>(initialPrior, pair_idx, labels, p0, N);
  gemm(p0, nullptr, fc1_w, fc1_b, h1, N, 128,    NCLS, 0);
  gemm(h1, nullptr, fc2_w, fc2_b, h2, N, 256,    128,  0);
  gemm(h2, nullptr, fc3_w, fc3_b, h3, N, 512,    256,  0);
  gemm(h3, nullptr, fc4_w, fc4_b, priorsN, N, D_MODEL, 512, 0);

  // 3) scatter into padded [frame][slot][D] layout
  hipMemsetAsync(X, 0, szMat, stream);
  hipMemsetAsync(P, 0, szMat, stream);
  scatter_kernel<<<N, 256, 0, stream>>>(features, priorsN, im_idx, pos, X, P, N);

  // 4) q/k/v projections (two chained D x D GEMMs each)
  const size_t DD = (size_t)D_MODEL * D_MODEL;
  gemm(X,  P,       Wq_w,             Wq_b,                 T1, MPAD, D_MODEL, D_MODEL, 0);
  gemm(T1, nullptr, in_proj_w,        in_proj_b,            Q,  MPAD, D_MODEL, D_MODEL, 0);
  gemm(X,  P,       Wk_w,             Wk_b,                 T1, MPAD, D_MODEL, D_MODEL, 0);
  gemm(T1, nullptr, in_proj_w + DD,   in_proj_b + D_MODEL,  Kb, MPAD, D_MODEL, D_MODEL, 0);
  gemm(X,  nullptr, Wv_w,             Wv_b,                 T1, MPAD, D_MODEL, D_MODEL, 0);
  gemm(T1, nullptr, in_proj_w + 2*DD, in_proj_b + 2*D_MODEL, V, MPAD, D_MODEL, D_MODEL, 0);

  // 5) attention + output projection
  attention_kernel<<<dim3(FRAMES, NHEAD), 256, 0, stream>>>(Q, Kb, V, counts, T1);
  gemm(T1, nullptr, out_proj_w, out_proj_b, T2, MPAD, D_MODEL, D_MODEL, 0);

  // 6) LN1, FFN, LN2
  resid_ln_kernel<<<MPAD, 256, 0, stream>>>(X, T2, ln1_g, ln1_b);
  gemm(X,  nullptr, lin1_w, lin1_b, Hf, MPAD, DFF,     D_MODEL, 1);
  gemm(Hf, nullptr, lin2_w, lin2_b, T1, MPAD, D_MODEL, DFF,     0);
  resid_ln_kernel<<<MPAD, 256, 0, stream>>>(X, T1, ln2_g, ln2_b);

  // 7) gather outputs: features then priors, concatenated
  gather_kernel<<<N, 256, 0, stream>>>(X, P, im_idx, pos, (float*)d_out, N);
}
